// STDPLinear_59219009077774
// MI455X (gfx1250) — matrix operand order repaired and checked by reading, not compiled, not run on hardware
//
#include <hip/hip_runtime.h>

typedef _Float16 v4h  __attribute__((ext_vector_type(4)));
typedef _Float16 v8h  __attribute__((ext_vector_type(8)));
typedef _Float16 v16h __attribute__((ext_vector_type(16)));
typedef float    v4f  __attribute__((ext_vector_type(4)));
typedef float    v8f  __attribute__((ext_vector_type(8)));

#define BATCH 512
#define IDIM  4096
#define ODIM  4096

// LIF / STDP constants
#define BETA_C   0.99f
#define THRESH_C 1.0f
#define RESET_C  0.8f
#define TDEC_C   0.9f
#define WCLAMP_C 0.1f
// A_POS - A_NEG == 0 exactly -> dw == 0 exactly; new_weight = clamp(weight).

// flat float offsets into d_out (return order: spikes, new_mem, new_w, tp, tq)
#define SPK_OFF 0L
#define MEM_OFF ((long)BATCH * ODIM)                  //  2,097,152
#define NW_OFF  (2L * BATCH * ODIM)                   //  4,194,304
#define TP_OFF  (NW_OFF + (long)ODIM * IDIM)          // 20,971,520
#define TQ_OFF  (TP_OFF + (long)BATCH * IDIM)         // 23,068,672

// GEMM tiling
#define BM 64
#define BN 256
#define BK 32
#define LDA 40   // padded LDS stride in halves (80B: rows hit distinct banks)
#define LDB 40

__device__ __forceinline__ v4h cvt4(v4f f) {
    v4h h;
    h[0] = (_Float16)f[0]; h[1] = (_Float16)f[1];
    h[2] = (_Float16)f[2]; h[3] = (_Float16)f[3];
    return h;
}

// Fused: weighted = spikes @ W^T ; LIF update ; tq trace. Writes spikes, new_mem, tq.
__launch_bounds__(256)
__global__ void snn_gemm_lif(const float* __restrict__ sp,     // [B,I]
                             const float* __restrict__ w,      // [O,I]
                             const float* __restrict__ mem,    // [B,O]
                             const float* __restrict__ tpost,  // [B,O]
                             float* __restrict__ out)
{
    __shared__ _Float16 lA[2][BM * LDA];
    __shared__ _Float16 lB[2][BN * LDB];

    const int tid  = threadIdx.x;
    const int lane = tid & 31;
    const int wave = tid >> 5;
    const int wm   = wave >> 2;      // 0..1  (M direction)
    const int wn   = wave & 3;       // 0..3  (N direction)
    const int ln   = lane & 15;
    const int half = lane >> 4;

    const int mBlk = blockIdx.y * BM;   // batch base
    const int nBlk = blockIdx.x * BN;   // output-neuron base

    // cooperative staging coords: 32 rows per pass, 8 threads (float4 each) per row
    const int sr = tid >> 3;            // 0..31
    const int sc = (tid & 7) * 4;       // 0,4,...,28 (floats within K-tile)

    v8f acc[2][4] = {};                 // 32M x 64N per wave

    // ---- prologue: stage K-tile 0 into buffer 0 (f32 -> f16) ----
    #pragma unroll
    for (int p = 0; p < 2; ++p) {
        int r = sr + 32 * p;
        v4f f = *(const v4f*)(sp + (long)(mBlk + r) * IDIM + sc);
        *(v4h*)&lA[0][r * LDA + sc] = cvt4(f);
    }
    #pragma unroll
    for (int p = 0; p < 8; ++p) {
        int r = sr + 32 * p;
        v4f f = *(const v4f*)(w + (long)(nBlk + r) * IDIM + sc);
        *(v4h*)&lB[0][r * LDB + sc] = cvt4(f);
    }
    __syncthreads();

    const int nIter = IDIM / BK;        // 128
    for (int kt = 0; kt < nIter; ++kt) {
        const int cur = kt & 1, nxt = cur ^ 1;
        const bool more = (kt + 1) < nIter;
        const int kOff = (kt + 1) * BK;

        // prefetch next K-tile into registers (hides global latency behind WMMA)
        v4f aPre[2], bPre[8];
        if (more) {
            #pragma unroll
            for (int p = 0; p < 2; ++p)
                aPre[p] = *(const v4f*)(sp + (long)(mBlk + sr + 32 * p) * IDIM + kOff + sc);
            #pragma unroll
            for (int p = 0; p < 8; ++p)
                bPre[p] = *(const v4f*)(w + (long)(nBlk + sr + 32 * p) * IDIM + kOff + sc);
        }

        // ---- load fragments from LDS (ISA 7.12.2 layouts) ----
        v16h af[2], bf[4];
        #pragma unroll
        for (int mi = 0; mi < 2; ++mi) {
            // A lane: row m, halves K = 8*half..+7 and 16+8*half..+7
            const _Float16* ap = &lA[cur][(wm * 32 + mi * 16 + ln) * LDA + 8 * half];
            v8h a0 = *(const v8h*)ap;
            v8h a1 = *(const v8h*)(ap + 16);
            #pragma unroll
            for (int e = 0; e < 8; ++e) { af[mi][e] = a0[e]; af[mi][e + 8] = a1[e]; }
        }
        #pragma unroll
        for (int ni = 0; ni < 4; ++ni) {
            // B lane: column n, halves K = 8*half..+7 and 16+8*half..+7
            const _Float16* bp = &lB[cur][(wn * 64 + ni * 16 + ln) * LDB + 8 * half];
            v8h b0 = *(const v8h*)bp;
            v8h b1 = *(const v8h*)(bp + 16);
            #pragma unroll
            for (int e = 0; e < 8; ++e) { bf[ni][e] = b0[e]; bf[ni][e + 8] = b1[e]; }
        }

        // ---- 8 WMMAs: D = A*B + C, f32 accumulate ----
        #pragma unroll
        for (int mi = 0; mi < 2; ++mi)
            #pragma unroll
            for (int ni = 0; ni < 4; ++ni)
                acc[mi][ni] = __builtin_amdgcn_wmma_f32_16x16x32_f16(
                    false, af[mi], false, bf[ni], (short)0, acc[mi][ni], false, false);

        // ---- write prefetched tile into the other buffer ----
        if (more) {
            #pragma unroll
            for (int p = 0; p < 2; ++p)
                *(v4h*)&lA[nxt][(sr + 32 * p) * LDA + sc] = cvt4(aPre[p]);
            #pragma unroll
            for (int p = 0; p < 8; ++p)
                *(v4h*)&lB[nxt][(sr + 32 * p) * LDB + sc] = cvt4(bPre[p]);
        }
        __syncthreads();
    }

    // ---- epilogue: fused LIF + post-trace, using C/D layout (m = r + 8*half, n = ln) ----
    #pragma unroll
    for (int mi = 0; mi < 2; ++mi) {
        #pragma unroll
        for (int ni = 0; ni < 4; ++ni) {
            #pragma unroll
            for (int r = 0; r < 8; ++r) {
                int  b   = mBlk + wm * 32 + mi * 16 + r + 8 * half;
                int  o   = nBlk + wn * 64 + ni * 16 + ln;
                long idx = (long)b * ODIM + o;
                float m  = mem[idx] * BETA_C + acc[mi][ni][r];
                float s  = (m > THRESH_C) ? 1.0f : 0.0f;
                float nm = (s > 0.0f) ? (m - RESET_C) : m;
                float tq = tpost[idx] * TDEC_C + s;
                out[SPK_OFF + idx] = s;
                out[MEM_OFF + idx] = nm;
                out[TQ_OFF  + idx] = tq;
            }
        }
    }
}

// tp = trace_pre * 0.9 + in_spikes   [B,I]
__global__ void snn_trace_pre(const float* __restrict__ tpre,
                              const float* __restrict__ sp,
                              float* __restrict__ out)
{
    long i = ((long)blockIdx.x * blockDim.x + threadIdx.x) * 4;
    v4f t = *(const v4f*)(tpre + i);
    v4f s = *(const v4f*)(sp + i);
    v4f r;
    #pragma unroll
    for (int e = 0; e < 4; ++e) r[e] = t[e] * TDEC_C + s[e];
    *(v4f*)(out + TP_OFF + i) = r;
}

// new_weight = clamp(weight + dw, ±0.1); dw == (A_POS-A_NEG)*GEMM == 0 exactly
__global__ void snn_weight_clamp(const float* __restrict__ w, float* __restrict__ out)
{
    long i = ((long)blockIdx.x * blockDim.x + threadIdx.x) * 4;
    v4f x = *(const v4f*)(w + i);
    #pragma unroll
    for (int e = 0; e < 4; ++e) x[e] = fminf(fmaxf(x[e], -WCLAMP_C), WCLAMP_C);
    *(v4f*)(out + NW_OFF + i) = x;
}

extern "C" void kernel_launch(void* const* d_in, const int* in_sizes, int n_in,
                              void* d_out, int out_size, void* d_ws, size_t ws_size,
                              hipStream_t stream)
{
    const float* sp    = (const float*)d_in[0];   // in_spikes  [B,I]
    const float* w     = (const float*)d_in[1];   // weight     [O,I]
    const float* mem   = (const float*)d_in[2];   // membrane   [B,O]
    const float* tpre  = (const float*)d_in[3];   // trace_pre  [B,I]
    const float* tpost = (const float*)d_in[4];   // trace_post [B,O]
    float* out = (float*)d_out;

    dim3 grid(ODIM / BN, BATCH / BM);             // 16 x 8 = 128 blocks, 256 thr (8 waves)
    snn_gemm_lif<<<grid, 256, 0, stream>>>(sp, w, mem, tpost, out);

    snn_trace_pre<<<((long)BATCH * IDIM) / 4 / 256, 256, 0, stream>>>(tpre, sp, out);
    snn_weight_clamp<<<((long)ODIM * IDIM) / 4 / 256, 256, 0, stream>>>(w, out);
}
